// AdaptiveTemporalKernels_51539607552833
// MI455X (gfx1250) — matrix operand order repaired and checked by reading, not compiled, not run on hardware
//
#include <hip/hip_runtime.h>
#include <hip/hip_bf16.h>
#include <math.h>

typedef _Float16 half16 __attribute__((ext_vector_type(16)));
typedef _Float16 half8  __attribute__((ext_vector_type(8)));
typedef float    float8 __attribute__((ext_vector_type(8)));

#define B_   8
#define S_   250
#define D_   128
#define E_   5120
#define H_   8
#define HD_  640
#define SP_  256
#define M_   (B_ * S_)      /* 2000 = 125 * 16 */
#define N1_  (3 * E_)       /* 15360 */

__device__ __forceinline__ half16 hcat(half8 a, half8 b) {
    return __builtin_shufflevector(a, b, 0,1,2,3,4,5,6,7,8,9,10,11,12,13,14,15);
}

// ---------------------------------------------------------------------------
// 1) g = mean_S(x)   (B,D)
// ---------------------------------------------------------------------------
__global__ void mean_kernel(const float* __restrict__ x, float* __restrict__ g) {
    int idx = blockIdx.x * blockDim.x + threadIdx.x;
    if (idx >= B_ * D_) return;
    int d = idx % D_, b = idx / D_;
    float s = 0.f;
    for (int t = 0; t < S_; ++t) s += x[((size_t)b * S_ + t) * D_ + d];
    g[idx] = s * (1.0f / S_);
}

// ---------------------------------------------------------------------------
// 2) h = gelu(g @ kg_w1^T + b1)   (B,D)
// ---------------------------------------------------------------------------
__global__ void kg1_kernel(const float* __restrict__ g, const float* __restrict__ w,
                           const float* __restrict__ bias, float* __restrict__ h) {
    int idx = blockIdx.x * blockDim.x + threadIdx.x;
    if (idx >= B_ * D_) return;
    int i = idx % D_, b = idx / D_;
    float s = bias[i];
    const float* gr = g + b * D_;
    const float* wr = w + (size_t)i * D_;
    for (int j = 0; j < D_; ++j) s += gr[j] * wr[j];
    h[idx] = 0.5f * s * (1.0f + erff(s * 0.70710678118654752f));
}

// ---------------------------------------------------------------------------
// 3) aw = tanh(h @ kg_w2^T + b2)   (B,E)
// ---------------------------------------------------------------------------
__global__ void kg2_kernel(const float* __restrict__ h, const float* __restrict__ w,
                           const float* __restrict__ bias, float* __restrict__ aw) {
    int idx = blockIdx.x * blockDim.x + threadIdx.x;
    if (idx >= B_ * E_) return;
    int e = idx % E_, b = idx / E_;
    float s = bias[e];
    const float* hr = h + b * D_;
    const float* wr = w + (size_t)e * D_;
    for (int j = 0; j < D_; ++j) s += hr[j] * wr[j];
    aw[idx] = tanhf(s);
}

// ---------------------------------------------------------------------------
// 4) depthwise dilated convs + adaptive weighting -> cat16 (B,S,E) fp16
// ---------------------------------------------------------------------------
__global__ void conv_cat_kernel(const float* __restrict__ x,
                                const float* __restrict__ w3,  const float* __restrict__ b3,
                                const float* __restrict__ w5,  const float* __restrict__ b5,
                                const float* __restrict__ w7,  const float* __restrict__ b7,
                                const float* __restrict__ w9,  const float* __restrict__ b9,
                                const float* __restrict__ w11, const float* __restrict__ b11,
                                const float* __restrict__ aw,
                                _Float16* __restrict__ cat) {
    size_t idx = (size_t)blockIdx.x * blockDim.x + threadIdx.x;
    if (idx >= (size_t)B_ * S_ * E_) return;
    int e = (int)(idx % E_);
    int s = (int)((idx / E_) % S_);
    int b = (int)(idx / ((size_t)E_ * S_));
    int ki = e / (8 * D_);
    int di = (e / D_) % 8;
    int d  = e % D_;
    const int ksz[5] = {3, 5, 7, 9, 11};
    const float* ws[5] = {w3, w5, w7, w9, w11};
    const float* bs[5] = {b3, b5, b7, b9, b11};
    int k = ksz[ki];
    const float* wp = ws[ki] + ((size_t)di * D_ + d) * k;
    float bias = bs[ki][di * D_ + d];
    int dil = di + 1;
    int pad = ((k - 1) * dil) / 2;
    float sum = 0.f;
    for (int t = 0; t < k; ++t) {
        int si = s - pad + t * dil;
        if (si >= 0 && si < S_)
            sum += x[((size_t)b * S_ + si) * D_ + d] * wp[t];
    }
    float v = (sum + bias) * aw[(size_t)b * E_ + e];
    cat[idx] = (_Float16)v;
}

// ---------------------------------------------------------------------------
// 5) fp32 -> fp16 weight conversion
// ---------------------------------------------------------------------------
__global__ void f32to16_kernel(const float* __restrict__ src, _Float16* __restrict__ dst,
                               size_t n) {
    size_t i = (size_t)blockIdx.x * blockDim.x + threadIdx.x;
    if (i < n) dst[i] = (_Float16)src[i];
}

// ---------------------------------------------------------------------------
// 6) generic WMMA GEMM:  C[M,N] = A16[M,K] @ W16[N,K]^T + bias[N]
//    block = 8 waves; wave tile = 32(M) x 64(N) (2 A frags x 4 B frags,
//    8 WMMAs per 32-wide K step -> 1.5 b128 loads per WMMA).
//    A fragment per lane (m = lane&15):  halves [ks..ks+15],
//    ks = 0 (lanes 0-15) / 16 (lanes 16-31), same K order as B.
//    B fragment per lane (n = lane&15):  16 contiguous K at ks = 0 / 16.
// ---------------------------------------------------------------------------
__global__ __launch_bounds__(256)
void wmma_gemm_kernel(const _Float16* __restrict__ A,
                      const _Float16* __restrict__ W,
                      const float* __restrict__ bias,
                      _Float16* __restrict__ C16,
                      float* __restrict__ C32,
                      int M, int N, int K) {
    int lane = threadIdx.x & 31;
    int wave = threadIdx.x >> 5;
    int mtiles = (M + 15) >> 4;                   // 16-row tiles (125)
    int wt = blockIdx.y * 8 + wave;               // 32-row wave tile
    int mt0 = wt * 2, mt1 = wt * 2 + 1;
    if (mt0 >= mtiles) return;                    // wave-uniform exit
    bool has1 = (mt1 < mtiles);
    int n0 = blockIdx.x * 64;
    int la = lane & 15;
    int ks_a = (lane < 16) ? 0 : 16;
    int ks_b = (lane < 16) ? 0 : 16;
    const _Float16* arow0 = A + (size_t)(mt0 * 16 + la) * K;
    const _Float16* arow1 = A + (size_t)((has1 ? mt1 : mt0) * 16 + la) * K;
    const _Float16* wr0 = W + (size_t)(n0 +  0 + la) * K + ks_b;
    const _Float16* wr1 = W + (size_t)(n0 + 16 + la) * K + ks_b;
    const _Float16* wr2 = W + (size_t)(n0 + 32 + la) * K + ks_b;
    const _Float16* wr3 = W + (size_t)(n0 + 48 + la) * K + ks_b;
    float8 acc00 = {}, acc01 = {}, acc02 = {}, acc03 = {};
    float8 acc10 = {}, acc11 = {}, acc12 = {}, acc13 = {};
    for (int k0 = 0; k0 < K; k0 += 32) {
        __builtin_prefetch(arow0 + k0 + 256, 0, 1);   // global_prefetch_b8
        __builtin_prefetch(arow1 + k0 + 256, 0, 1);
        half16 a0 = hcat(*(const half8*)(arow0 + k0 + ks_a),
                         *(const half8*)(arow0 + k0 + ks_a + 8));
        half16 a1 = hcat(*(const half8*)(arow1 + k0 + ks_a),
                         *(const half8*)(arow1 + k0 + ks_a + 8));
        half16 b0 = hcat(*(const half8*)(wr0 + k0), *(const half8*)(wr0 + k0 + 8));
        half16 b1 = hcat(*(const half8*)(wr1 + k0), *(const half8*)(wr1 + k0 + 8));
        half16 b2 = hcat(*(const half8*)(wr2 + k0), *(const half8*)(wr2 + k0 + 8));
        half16 b3 = hcat(*(const half8*)(wr3 + k0), *(const half8*)(wr3 + k0 + 8));
        acc00 = __builtin_amdgcn_wmma_f32_16x16x32_f16(false, a0, false, b0, (short)0, acc00, false, false);
        acc10 = __builtin_amdgcn_wmma_f32_16x16x32_f16(false, a1, false, b0, (short)0, acc10, false, false);
        acc01 = __builtin_amdgcn_wmma_f32_16x16x32_f16(false, a0, false, b1, (short)0, acc01, false, false);
        acc11 = __builtin_amdgcn_wmma_f32_16x16x32_f16(false, a1, false, b1, (short)0, acc11, false, false);
        acc02 = __builtin_amdgcn_wmma_f32_16x16x32_f16(false, a0, false, b2, (short)0, acc02, false, false);
        acc12 = __builtin_amdgcn_wmma_f32_16x16x32_f16(false, a1, false, b2, (short)0, acc12, false, false);
        acc03 = __builtin_amdgcn_wmma_f32_16x16x32_f16(false, a0, false, b3, (short)0, acc03, false, false);
        acc13 = __builtin_amdgcn_wmma_f32_16x16x32_f16(false, a1, false, b3, (short)0, acc13, false, false);
    }
    // C layout: VGPR r -> (m = (lane>=16)*8 + r, n = lane&15)
    int m0 = (lane >> 4) * 8;
    float8 t0[4] = {acc00, acc01, acc02, acc03};
    float8 t1[4] = {acc10, acc11, acc12, acc13};
    for (int j = 0; j < 4; ++j) {
        int ncol = n0 + j * 16 + la;
        float bv = bias ? bias[ncol] : 0.0f;
        for (int r = 0; r < 8; ++r) {
            float v = t0[j][r] + bv;
            size_t off = (size_t)(mt0 * 16 + m0 + r) * N + ncol;
            if (C32) C32[off] = v;
            else     C16[off] = (_Float16)v;
        }
        if (has1) {
            for (int r = 0; r < 8; ++r) {
                float v = t1[j][r] + bv;
                size_t off = (size_t)(mt1 * 16 + m0 + r) * N + ncol;
                if (C32) C32[off] = v;
                else     C16[off] = (_Float16)v;
            }
        }
    }
}

// ---------------------------------------------------------------------------
// 7) pack qkv16 (B,S,3E) -> qp/kp (b,h,256,640) zero-padded, vT (b,h,640,256)
// ---------------------------------------------------------------------------
__global__ void pack_qkv_kernel(const _Float16* __restrict__ qkv,
                                _Float16* __restrict__ qp,
                                _Float16* __restrict__ kp,
                                _Float16* __restrict__ vT) {
    size_t idx = (size_t)blockIdx.x * blockDim.x + threadIdx.x;
    if (idx >= (size_t)B_ * H_ * SP_ * HD_) return;
    int c = (int)(idx % HD_);
    int s = (int)((idx / HD_) % SP_);
    int h = (int)((idx / ((size_t)HD_ * SP_)) % H_);
    int b = (int)(idx / ((size_t)HD_ * SP_ * H_));
    _Float16 qv = (_Float16)0.f, kv = (_Float16)0.f, vv = (_Float16)0.f;
    if (s < S_) {
        size_t base = ((size_t)b * S_ + s) * (size_t)N1_ + (size_t)h * HD_ + c;
        qv = qkv[base];
        kv = qkv[base + E_];
        vv = qkv[base + 2 * E_];
    }
    size_t bh = (size_t)b * H_ + h;
    qp[(bh * SP_ + s) * HD_ + c] = qv;
    kp[(bh * SP_ + s) * HD_ + c] = kv;
    vT[(bh * HD_ + c) * SP_ + s] = vv;
}

// ---------------------------------------------------------------------------
// 8) attention: one wave per (b, h, 16-row q tile). Scores & att@v via WMMA.
// ---------------------------------------------------------------------------
__global__ __launch_bounds__(32)
void attn_kernel(const _Float16* __restrict__ qp,
                 const _Float16* __restrict__ kp,
                 const _Float16* __restrict__ vT,
                 _Float16* __restrict__ ao) {
    __shared__ float    sc[16][SP_];
    __shared__ _Float16 at16[16][SP_];
    int lane  = threadIdx.x;
    int qtile = blockIdx.x & 15;
    int h     = (blockIdx.x >> 4) & 7;
    int b     = blockIdx.x >> 7;
    int la    = lane & 15;
    int ks_a  = (lane < 16) ? 0 : 16;
    int ks_b  = (lane < 16) ? 0 : 16;
    size_t bh = (size_t)b * H_ + h;
    const _Float16* qrow = qp + (bh * SP_ + (size_t)(qtile * 16 + la)) * HD_;
    const _Float16* kbas = kp + bh * SP_ * HD_;
    const _Float16* vtb  = vT + bh * HD_ * SP_;
    const float scale = 0.039528470752104741f;   // 1/sqrt(640)

    // scores strip: 16 x 256 into LDS
    for (int j = 0; j < 16; ++j) {
        float8 acc = {};
        const _Float16* krow = kbas + (size_t)(j * 16 + la) * HD_;
        for (int c0 = 0; c0 < HD_; c0 += 32) {
            half16 af = hcat(*(const half8*)(qrow + c0 + ks_a),
                             *(const half8*)(qrow + c0 + ks_a + 8));
            half16 bf = hcat(*(const half8*)(krow + c0 + ks_b),
                             *(const half8*)(krow + c0 + ks_b + 8));
            acc = __builtin_amdgcn_wmma_f32_16x16x32_f16(false, af, false, bf, (short)0, acc, false, false);
        }
        int m0 = (lane >> 4) * 8;
        for (int r = 0; r < 8; ++r) sc[m0 + r][j * 16 + la] = acc[r] * scale;
    }
    __syncthreads();

    // masked softmax (cols >= 250 excluded), rewrite as fp16 probs
    if (lane < 16) {
        float mx = -1e30f;
        for (int c = 0; c < S_; ++c) mx = fmaxf(mx, sc[lane][c]);
        float sum = 0.f;
        for (int c = 0; c < S_; ++c) { float e = expf(sc[lane][c] - mx); sc[lane][c] = e; sum += e; }
        float inv = 1.0f / sum;
        for (int c = 0; c < SP_; ++c)
            at16[lane][c] = (c < S_) ? (_Float16)(sc[lane][c] * inv) : (_Float16)0.f;
    }
    __syncthreads();

    // ao = att(16x256) @ v(256x640): A fragments from LDS (ds_load_b128)
    for (int nt = 0; nt < HD_ / 16; ++nt) {
        float8 acc = {};
        const _Float16* vrow = vtb + (size_t)(nt * 16 + la) * SP_;
        for (int c0 = 0; c0 < SP_; c0 += 32) {
            half16 af = hcat(*(const half8*)(&at16[la][c0 + ks_a]),
                             *(const half8*)(&at16[la][c0 + ks_a + 8]));
            half16 bf = hcat(*(const half8*)(vrow + c0 + ks_b),
                             *(const half8*)(vrow + c0 + ks_b + 8));
            acc = __builtin_amdgcn_wmma_f32_16x16x32_f16(false, af, false, bf, (short)0, acc, false, false);
        }
        int m0 = (lane >> 4) * 8;
        for (int r = 0; r < 8; ++r) {
            int s = qtile * 16 + m0 + r;
            if (s < S_)
                ao[((size_t)b * S_ + s) * E_ + (size_t)h * HD_ + nt * 16 + la] = (_Float16)acc[r];
        }
    }
}

// ---------------------------------------------------------------------------
// 9) residual + LayerNorm epilogue (block = 128 threads = one (b,s) row)
// ---------------------------------------------------------------------------
__global__ void ln_kernel(const float* __restrict__ p, const float* __restrict__ x,
                          const float* __restrict__ gamma, const float* __restrict__ beta,
                          float* __restrict__ out) {
    __shared__ float red[128];
    int row = blockIdx.x, d = threadIdx.x;
    float v = p[(size_t)row * D_ + d] + x[(size_t)row * D_ + d];
    red[d] = v; __syncthreads();
    for (int s = 64; s > 0; s >>= 1) { if (d < s) red[d] += red[d + s]; __syncthreads(); }
    float mu = red[0] * (1.0f / D_); __syncthreads();
    float c = v - mu;
    red[d] = c * c; __syncthreads();
    for (int s = 64; s > 0; s >>= 1) { if (d < s) red[d] += red[d + s]; __syncthreads(); }
    float var = red[0] * (1.0f / D_);
    out[(size_t)row * D_ + d] = c * rsqrtf(var + 1e-5f) * gamma[d] + beta[d];
}

// ---------------------------------------------------------------------------
extern "C" void kernel_launch(void* const* d_in, const int* in_sizes, int n_in,
                              void* d_out, int out_size, void* d_ws, size_t ws_size,
                              hipStream_t stream) {
    const float* x          = (const float*)d_in[0];
    const float* kg_w1      = (const float*)d_in[1];
    const float* kg_b1      = (const float*)d_in[2];
    const float* kg_w2      = (const float*)d_in[3];
    const float* kg_b2      = (const float*)d_in[4];
    const float* cw[5], *cb[5];
    for (int i = 0; i < 5; ++i) { cw[i] = (const float*)d_in[5 + 2 * i]; cb[i] = (const float*)d_in[6 + 2 * i]; }
    const float* attn_in_w  = (const float*)d_in[15];
    const float* attn_in_b  = (const float*)d_in[16];
    const float* attn_out_w = (const float*)d_in[17];
    const float* attn_out_b = (const float*)d_in[18];
    const float* proj_w     = (const float*)d_in[19];
    const float* proj_b     = (const float*)d_in[20];
    const float* gamma      = (const float*)d_in[21];
    const float* beta       = (const float*)d_in[22];
    float* out = (float*)d_out;

    size_t off = 0;
    auto alloc = [&](size_t bytes) -> void* {
        void* p = (char*)d_ws + off;
        off += (bytes + 255) & ~(size_t)255;
        return p;
    };
    float*    g      = (float*)alloc((size_t)B_ * D_ * 4);
    float*    hbuf   = (float*)alloc((size_t)B_ * D_ * 4);
    float*    awbuf  = (float*)alloc((size_t)B_ * E_ * 4);
    _Float16* cat16  = (_Float16*)alloc((size_t)M_ * E_ * 2);
    _Float16* w1h    = (_Float16*)alloc((size_t)N1_ * E_ * 2);
    _Float16* w2h    = (_Float16*)alloc((size_t)E_ * E_ * 2);
    _Float16* w3h    = (_Float16*)alloc((size_t)D_ * E_ * 2);
    _Float16* qkv16  = (_Float16*)alloc((size_t)M_ * N1_ * 2);
    _Float16* qp     = (_Float16*)alloc((size_t)B_ * H_ * SP_ * HD_ * 2);
    _Float16* kpbuf  = (_Float16*)alloc((size_t)B_ * H_ * SP_ * HD_ * 2);
    _Float16* vT     = (_Float16*)alloc((size_t)B_ * H_ * SP_ * HD_ * 2);
    _Float16* ao16   = (_Float16*)alloc((size_t)M_ * E_ * 2);
    _Float16* y16    = (_Float16*)alloc((size_t)M_ * E_ * 2);
    float*    pout   = (float*)alloc((size_t)M_ * D_ * 4);

    // prologue: gates
    mean_kernel<<<(B_ * D_ + 255) / 256, 256, 0, stream>>>(x, g);
    kg1_kernel<<<(B_ * D_ + 255) / 256, 256, 0, stream>>>(g, kg_w1, kg_b1, hbuf);
    kg2_kernel<<<(B_ * E_ + 255) / 256, 256, 0, stream>>>(hbuf, kg_w2, kg_b2, awbuf);

    // convs -> cat16
    {
        size_t total = (size_t)B_ * S_ * E_;
        conv_cat_kernel<<<(unsigned)((total + 255) / 256), 256, 0, stream>>>(
            x, cw[0], cb[0], cw[1], cb[1], cw[2], cb[2], cw[3], cb[3], cw[4], cb[4],
            awbuf, cat16);
    }

    // fp16 weights
    {
        size_t n1 = (size_t)N1_ * E_, n2 = (size_t)E_ * E_, n3 = (size_t)D_ * E_;
        f32to16_kernel<<<(unsigned)((n1 + 255) / 256), 256, 0, stream>>>(attn_in_w,  w1h, n1);
        f32to16_kernel<<<(unsigned)((n2 + 255) / 256), 256, 0, stream>>>(attn_out_w, w2h, n2);
        f32to16_kernel<<<(unsigned)((n3 + 255) / 256), 256, 0, stream>>>(proj_w,     w3h, n3);
    }

    // wave tiles: 32 rows each; 63 tiles over M=2000; 8 waves per block
    int wtiles = ((M_ / 16) + 1) / 2;               // 63
    int gy = (wtiles + 7) / 8;                      // 8

    dim3 blk(256);
    dim3 gq(N1_ / 64, gy);
    wmma_gemm_kernel<<<gq, blk, 0, stream>>>(cat16, w1h, attn_in_b, qkv16, nullptr, M_, N1_, E_);

    {
        size_t total = (size_t)B_ * H_ * SP_ * HD_;
        pack_qkv_kernel<<<(unsigned)((total + 255) / 256), 256, 0, stream>>>(qkv16, qp, kpbuf, vT);
    }

    attn_kernel<<<B_ * H_ * 16, 32, 0, stream>>>(qp, kpbuf, vT, ao16);

    dim3 go(E_ / 64, gy);
    wmma_gemm_kernel<<<go, blk, 0, stream>>>(ao16, w2h, attn_out_b, y16, nullptr, M_, E_, E_);

    dim3 gp(D_ / 64, gy);
    wmma_gemm_kernel<<<gp, blk, 0, stream>>>(y16, w3h, proj_b, nullptr, pout, M_, D_, E_);

    ln_kernel<<<M_, 128, 0, stream>>>(pout, x, gamma, beta, out);
}
